// DeformableTransformer_20169166422039
// MI455X (gfx1250) — matrix operand order repaired and checked by reading, not compiled, not run on hardware
//
#include <hip/hip_runtime.h>
#include <math.h>

// ---------------- problem constants ----------------
#define B_  2
#define S_  13294
#define L_  13294          // Lq == S
#define D_  256
#define NH_ 8
#define NL_ 4
#define NP_ 4
#define HD_ 32             // D / NH

// ---------------- types ----------------
typedef unsigned short u16;
typedef __bf16 bf16_t;
typedef bf16_t v16bf __attribute__((ext_vector_type(16)));
typedef u16    v16u  __attribute__((ext_vector_type(16)));
typedef float  v8f   __attribute__((ext_vector_type(8)));

__device__ __forceinline__ u16 f2bf_rne(float f) {
  unsigned u = __float_as_uint(f);
  unsigned r = u + 0x7fffu + ((u >> 16) & 1u);   // round-to-nearest-even
  return (u16)(r >> 16);
}
__device__ __forceinline__ float bf2f(u16 s) {
  return __uint_as_float(((unsigned)s) << 16);
}
__device__ __forceinline__ void split_bf(float f, u16& h, u16& l) {
  u16 hh = f2bf_rne(f);
  h = hh;
  l = f2bf_rne(f - bf2f(hh));                    // residual for fp32-class accuracy
}

// ---------------- WMMA GEMM:  C[M,N] = A[M,256] * Bw[256,N] + bias[N] ----------------
// bf16 hi/lo split (3 wmma per K=32) -> ~fp32 accuracy on v_wmma_f32_16x16x32_bf16.
#define TM 32
#define TN 64
#define TK 64
#define LDK 72   // padded row stride in u16 elements

template <int N>
__global__ __launch_bounds__(256) void wmma_gemm_bias(
    const float* __restrict__ A, const float* __restrict__ Bw,
    const float* __restrict__ bias, float* __restrict__ C,
    int M)
{
  __shared__ u16 AsH[TM][LDK];
  __shared__ u16 AsL[TM][LDK];
  __shared__ u16 BsH[TN][LDK];   // transposed: [n][k]
  __shared__ u16 BsL[TN][LDK];

  const int K = D_;
  const int tid  = threadIdx.x;
  const int lane = tid & 31;
  const int wave = tid >> 5;       // 0..7
  const int wm   = wave & 1;       // 2 m-tiles of 16
  const int wn   = wave >> 1;      // 4 n-tiles of 16
  const int block_m = blockIdx.x * TM;
  const int block_n = blockIdx.y * TN;

  v8f acc = {};

  const int m   = lane & 15;
  const int alo = (lane >> 4) * 16;   // A frag K sub-base
  const int bko = (lane >> 4) * 16;   // B frag K base
  const int n   = lane & 15;

  for (int k0 = 0; k0 < K; k0 += TK) {
    // ---- prefetch next K-chunk while this one is staged/computed ----
    if (k0 + TK < K) {
      int pr = block_m + (tid & (TM - 1)); if (pr >= M) pr = M - 1;
      __builtin_prefetch(A + (size_t)pr * K + k0 + TK, 0, 3);
      int br = k0 + TK + (tid & (TK - 1));
      __builtin_prefetch(Bw + (size_t)br * N + block_n, 0, 3);
    }
    // ---- stage A tile (TM x TK floats), hi/lo bf16 split, coalesced float4 ----
    for (int it = tid; it < (TM * TK) / 4; it += 256) {
      int r  = it >> 4;              // TK/4 == 16
      int c4 = (it & 15) << 2;
      int gr = block_m + r; if (gr >= M) gr = M - 1;
      const float4 v = *(const float4*)(A + (size_t)gr * K + k0 + c4);
      split_bf(v.x, AsH[r][c4 + 0], AsL[r][c4 + 0]);
      split_bf(v.y, AsH[r][c4 + 1], AsL[r][c4 + 1]);
      split_bf(v.z, AsH[r][c4 + 2], AsL[r][c4 + 2]);
      split_bf(v.w, AsH[r][c4 + 3], AsL[r][c4 + 3]);
    }
    // ---- stage B tile (TK x TN floats) transposed to [n][k] ----
    for (int it = tid; it < TK * (TN / 4); it += 256) {
      int kr  = it >> 4;             // TN/4 == 16
      int nc4 = (it & 15) << 2;
      const float4 v = *(const float4*)(Bw + (size_t)(k0 + kr) * N + block_n + nc4);
      split_bf(v.x, BsH[nc4 + 0][kr], BsL[nc4 + 0][kr]);
      split_bf(v.y, BsH[nc4 + 1][kr], BsL[nc4 + 1][kr]);
      split_bf(v.z, BsH[nc4 + 2][kr], BsL[nc4 + 2][kr]);
      split_bf(v.w, BsH[nc4 + 3][kr], BsL[nc4 + 3][kr]);
    }
    __syncthreads();

    // ---- 2 K-slices of 32, 3 WMMAs each (hi*hi + hi*lo + lo*hi) ----
#pragma unroll
    for (int kc = 0; kc < TK; kc += 32) {
      v16u ahu, alu, bhu, blu;
      const int ar = wm * 16 + m;
#pragma unroll
      for (int j = 0; j < 8; ++j) {
        ahu[j]     = AsH[ar][kc + alo + j];
        ahu[j + 8] = AsH[ar][kc + 8 + alo + j];
        alu[j]     = AsL[ar][kc + alo + j];
        alu[j + 8] = AsL[ar][kc + 8 + alo + j];
      }
      const int br = wn * 16 + n;
#pragma unroll
      for (int j = 0; j < 16; ++j) {
        bhu[j] = BsH[br][kc + bko + j];
        blu[j] = BsL[br][kc + bko + j];
      }
      v16bf ah = __builtin_bit_cast(v16bf, ahu);
      v16bf al = __builtin_bit_cast(v16bf, alu);
      v16bf bh = __builtin_bit_cast(v16bf, bhu);
      v16bf bl = __builtin_bit_cast(v16bf, blu);

      acc = __builtin_amdgcn_wmma_f32_16x16x32_bf16(false, ah, false, bh, (short)0, acc, false, false);
      acc = __builtin_amdgcn_wmma_f32_16x16x32_bf16(false, ah, false, bl, (short)0, acc, false, false);
      acc = __builtin_amdgcn_wmma_f32_16x16x32_bf16(false, al, false, bh, (short)0, acc, false, false);
    }
    __syncthreads();
  }

  // ---- epilogue: fused bias; C/D layout: VGPR r -> rows r (lanes 0-15) / r+8 (lanes 16-31)
  // N is compile-time -> row stride is a shift; fast path is block-uniform (no exec churn).
  const int gn = block_n + wn * 16 + (lane & 15);
  const int mrow_base = block_m + wm * 16 + ((lane >> 4) * 8);
  const float bn = bias[gn];
  float* cp = C + (size_t)mrow_base * N + gn;
  if (block_m + TM <= M) {
#pragma unroll
    for (int r = 0; r < 8; ++r) { *cp = acc[r] + bn; cp += N; }
  } else {
#pragma unroll
    for (int r = 0; r < 8; ++r) {
      if (mrow_base + r < M) *cp = acc[r] + bn;
      cp += N;
    }
  }
}

// ---------------- softmax + bilinear deformable sampling ----------------
// One wave32 per (b, q, h); lane == channel (HD = 32). Corner reads are one
// coalesced 128B global_load_b32 per wave. All control flow is wave-uniform.
__global__ __launch_bounds__(256) void deform_sample(
    const float* __restrict__ valp,   // [B, S, D]  (col = h*HD + c)
    const float* __restrict__ offs,   // [B, L, NH, NL, NP, 2]
    const float* __restrict__ attn,   // [B, L, NH, 16]
    const float* __restrict__ refp,   // [B, L, NL, 2]
    float* __restrict__ outacc)       // [B, L, D]
{
  const int gid  = blockIdx.x * blockDim.x + threadIdx.x;
  const int w    = gid >> 5;
  const int lane = gid & 31;
  if (w >= B_ * L_ * NH_) return;
  const int h  = w & 7;        // NH = 8
  const int qb = w >> 3;       // = b*L + q
  const int b  = qb / L_;

  // softmax over 16 logits (redundant per lane; 64B broadcast load)
  const float* aw = attn + (size_t)w * 16;
  float lg[16];
#pragma unroll
  for (int i = 0; i < 16; ++i) lg[i] = aw[i];
  float mx = lg[0];
#pragma unroll
  for (int i = 1; i < 16; ++i) mx = fmaxf(mx, lg[i]);
  float ssum = 0.f;
#pragma unroll
  for (int i = 0; i < 16; ++i) { lg[i] = __expf(lg[i] - mx); ssum += lg[i]; }
  const float inv = 1.0f / ssum;

  const float* off = offs + (size_t)w * 32;             // NL*NP*2 floats
  const float* rp  = refp + (size_t)qb * (NL_ * 2);

  const int Hs[4] = {100, 50, 25, 13};
  const int Ws[4] = {100, 50, 25, 13};
  const int st[4] = {0, 10000, 12500, 13125};

  float accv = 0.f;
#pragma unroll
  for (int lvl = 0; lvl < NL_; ++lvl) {
    const int Wl = Ws[lvl], Hl = Hs[lvl];
    const float rx = rp[lvl * 2 + 0];
    const float ry = rp[lvl * 2 + 1];
    const float* vbase = valp + ((size_t)(b * S_ + st[lvl])) * D_ + h * HD_ + lane;
#pragma unroll
    for (int p = 0; p < NP_; ++p) {
      const float ox = off[(lvl * NP_ + p) * 2 + 0];
      const float oy = off[(lvl * NP_ + p) * 2 + 1];
      // (rx + ox/W)*W - 0.5 == rx*W + ox - 0.5
      const float x = rx * (float)Wl + ox - 0.5f;
      const float y = ry * (float)Hl + oy - 0.5f;
      const float x0f = floorf(x), y0f = floorf(y);
      const int x0 = (int)x0f, y0 = (int)y0f;
      const float wx1 = x - x0f, wx0 = 1.f - wx1;
      const float wy1 = y - y0f, wy0 = 1.f - wy1;

      const bool xin0 = (x0 >= 0) && (x0 < Wl);
      const bool xin1 = (x0 + 1 >= 0) && (x0 + 1 < Wl);
      const bool yin0 = (y0 >= 0) && (y0 < Hl);
      const bool yin1 = (y0 + 1 >= 0) && (y0 + 1 < Hl);

      float v00 = 0.f, v10 = 0.f, v01 = 0.f, v11 = 0.f;
      const float* row0 = vbase + (ptrdiff_t)y0 * Wl * D_;
      const float* row1 = row0 + (ptrdiff_t)Wl * D_;
      if (yin0) {
        if (xin0) v00 = row0[(ptrdiff_t)x0 * D_];
        if (xin1) v10 = row0[(ptrdiff_t)(x0 + 1) * D_];
      }
      if (yin1) {
        if (xin0) v01 = row1[(ptrdiff_t)x0 * D_];
        if (xin1) v11 = row1[(ptrdiff_t)(x0 + 1) * D_];
      }
      const float s = wy0 * (wx0 * v00 + wx1 * v10) + wy1 * (wx0 * v01 + wx1 * v11);
      accv = fmaf(lg[lvl * NP_ + p] * inv, s, accv);
    }
  }
  outacc[(size_t)w * 32 + lane] = accv;
}

// ---------------- launcher ----------------
extern "C" void kernel_launch(void* const* d_in, const int* in_sizes, int n_in,
                              void* d_out, int out_size, void* d_ws, size_t ws_size,
                              hipStream_t stream) {
  const float* query  = (const float*)d_in[0];
  const float* refp   = (const float*)d_in[1];
  const float* value  = (const float*)d_in[2];
  const float* W_val  = (const float*)d_in[3];
  const float* b_val  = (const float*)d_in[4];
  const float* W_off  = (const float*)d_in[5];
  const float* b_off  = (const float*)d_in[6];
  const float* W_attn = (const float*)d_in[7];
  const float* b_attn = (const float*)d_in[8];
  const float* W_out  = (const float*)d_in[9];
  const float* b_out  = (const float*)d_in[10];
  float* out = (float*)d_out;

  const int M = B_ * S_;                       // 26588 rows for every GEMM
  size_t o = 0;
  float* valp   = (float*)((char*)d_ws + o); o += (size_t)M * D_ * sizeof(float);
  float* offsb  = (float*)((char*)d_ws + o); o += (size_t)M * D_ * sizeof(float);
  float* attnb  = (float*)((char*)d_ws + o); o += (size_t)M * 128 * sizeof(float);
  float* outacc = (float*)((char*)d_ws + o); o += (size_t)M * D_ * sizeof(float);
  (void)ws_size; (void)in_sizes; (void)n_in; (void)out_size;

  dim3 blk(256);
  dim3 g256((M + TM - 1) / TM, 256 / TN);
  dim3 g128((M + TM - 1) / TM, 128 / TN);

  // 1) value projection
  wmma_gemm_bias<256><<<g256, blk, 0, stream>>>(value, W_val, b_val, valp, M);
  // 2) sampling offsets
  wmma_gemm_bias<256><<<g256, blk, 0, stream>>>(query, W_off, b_off, offsb, M);
  // 3) attention logits
  wmma_gemm_bias<128><<<g128, blk, 0, stream>>>(query, W_attn, b_attn, attnb, M);
  // 4) softmax + bilinear sampling + weighted accumulate
  const int threads = B_ * L_ * NH_ * 32;
  deform_sample<<<(threads + 255) / 256, blk, 0, stream>>>(valp, offsb, attnb, refp, outacc);
  // 5) output projection
  wmma_gemm_bias<256><<<g256, blk, 0, stream>>>(outacc, W_out, b_out, out, M);
}
